// Rigid_MultiHeadAttention_20401094656149
// MI455X (gfx1250) — matrix operand order repaired and checked by reading, not compiled, not run on hardware
//
#include <hip/hip_runtime.h>

// Rigid multi-head attention for MI455X (gfx1250, wave32).
// Heavy GEMMs use v_wmma_f32_16x16x32_bf16 (fp32 accumulate); bf16 tile
// staging uses CDNA5 async global->LDS copies (ASYNCcnt-tracked).

#define SEQ    1024
#define BATCH  2
#define DMODEL 768
#define NHEADS 12
#define HDIM   64
#define MTOT   (BATCH * SEQ)

typedef __attribute__((ext_vector_type(16))) __bf16 v16bf;
typedef __attribute__((ext_vector_type(8)))  float  v8f;

struct __align__(16) U128 { unsigned int x[4]; };
struct __align__(8)  U64  { unsigned int x[2]; };
struct __align__(16) F128 { float x, y, z, w; };

union FragU { v16bf v; unsigned short s[16]; U128 q[2]; };

__device__ __forceinline__ unsigned short f2bf(float f) {
  unsigned int u = __float_as_uint(f);
  u += 0x7FFFu + ((u >> 16) & 1u);         // round-to-nearest-even truncation
  return (unsigned short)(u >> 16);
}

// CDNA5 async global->LDS copy, 8 bytes per lane (GLOBAL_LOAD_ASYNC_TO_LDS_B64).
// LDS aperture: flat addr[31:0] == LDS byte offset (ISA 10.2).
__device__ __forceinline__ void async_copy_b64(void* lds_dst, const void* gsrc) {
  unsigned lo = (unsigned)(unsigned long long)lds_dst;
  asm volatile("global_load_async_to_lds_b64 %0, %1, off"
               :: "v"(lo), "v"(gsrc)
               : "memory");
}
__device__ __forceinline__ void wait_async() {
  asm volatile("s_wait_asynccnt 0x0" ::: "memory");
}

// A-fragment (16x32 bf16, M=lane&15 for both lane halves):
//   elems 0..7  -> K = 8*khalf + 0..7
//   elems 8..15 -> K = 16 + 8*khalf + 0..7
__device__ __forceinline__ v16bf frag_a_lds(const unsigned short* rowbase, int kbase, int khalf) {
  FragU u;
  const unsigned short* p = rowbase + kbase + 8 * khalf;
  u.q[0] = *reinterpret_cast<const U128*>(p);
  u.q[1] = *reinterpret_cast<const U128*>(p + 16);
  return u.v;
}
// B-fragment (32x16 bf16, N=lane&15): elems 0..7 -> K = 8*khalf + 0..7, elems 8..15 -> K = 16 + 8*khalf + 0..7
__device__ __forceinline__ v16bf frag_b_lds(const unsigned short* rowbase, int kbase, int khalf) {
  FragU u;
  const unsigned short* p = rowbase + kbase + 8 * khalf;
  u.q[0] = *reinterpret_cast<const U128*>(p);
  u.q[1] = *reinterpret_cast<const U128*>(p + 16);
  return u.v;
}

__device__ __forceinline__ v8f wmma_bf16(v16bf a, v16bf b, v8f c) {
  return __builtin_amdgcn_wmma_f32_16x16x32_bf16(false, a, false, b, (short)0, c, false, false);
}

// ---------------------------------------------------------------------------
// Generic NT GEMM: C[z] = alpha * A[z] (MxK) * B[z]^T (B is NxK row-major)
//                + bias[n] + optional rank-3 tail (frame @ tailB[:,K:K+3]^T)
// Block tile 128x64, BK=64, 8 waves of 32x32 (2x2 WMMA tiles, 8 WMMA/stage).
// f32 sources convert to bf16 in the staging path; bf16 sources use async
// global->LDS copies (no VGPR round trip).
// ---------------------------------------------------------------------------
template <int A_F32, int B_F32, int OUT_BF16>
__global__ __launch_bounds__(256) void gemm_nt_kernel(
    const void* __restrict__ Av, int lda, long long sA,
    const void* __restrict__ Bv, int ldb, long long sB,
    void* __restrict__ Cv, int ldc, long long sC,
    int M, int N, int K, float alpha,
    const float* __restrict__ bias,
    const float* __restrict__ frame,    // [M,3] or nullptr
    const float* __restrict__ tailB) {  // f32, same ldb, cols K..K+2
  __shared__ unsigned short As[128 * 64];   // 16 KB
  __shared__ unsigned short Bs[64 * 64];    //  8 KB

  const int t = threadIdx.x;
  const int lane = t & 31, w = t >> 5;        // wave32
  const int lrow = lane & 15, khalf = lane >> 4;
  const int wrow = w >> 1, wcol = w & 1;      // 4x2 wave grid -> 32x32 per wave
  const int bm = blockIdx.y * 128, bn = blockIdx.x * 64;
  const long long z = blockIdx.z;

  v8f acc[2][2];
#pragma unroll
  for (int i = 0; i < 2; ++i)
#pragma unroll
    for (int j = 0; j < 2; ++j)
#pragma unroll
      for (int r = 0; r < 8; ++r) acc[i][j][r] = 0.0f;

  for (int kb = 0; kb < K; kb += 64) {
    // stage A tile 128x64 (k-contiguous in LDS)
    for (int c = t; c < 2048; c += 256) {
      int row = c >> 4, col = (c & 15) * 4;
      unsigned short* d = &As[row * 64 + col];
      long long gr = bm + row;
      if (A_F32) {
        const float* src = (const float*)Av + z * sA + gr * (long long)lda + kb + col;
        F128 f = *reinterpret_cast<const F128*>(src);
        d[0] = f2bf(f.x); d[1] = f2bf(f.y); d[2] = f2bf(f.z); d[3] = f2bf(f.w);
      } else {
        const unsigned short* src =
            (const unsigned short*)Av + z * sA + gr * (long long)lda + kb + col;
        async_copy_b64(d, src);
      }
    }
    // stage B tile 64x64
    for (int c = t; c < 1024; c += 256) {
      int row = c >> 4, col = (c & 15) * 4;
      unsigned short* d = &Bs[row * 64 + col];
      long long gr = bn + row;
      if (B_F32) {
        const float* src = (const float*)Bv + z * sB + gr * (long long)ldb + kb + col;
        d[0] = f2bf(src[0]); d[1] = f2bf(src[1]); d[2] = f2bf(src[2]); d[3] = f2bf(src[3]);
      } else {
        const unsigned short* src =
            (const unsigned short*)Bv + z * sB + gr * (long long)ldb + kb + col;
        async_copy_b64(d, src);
      }
    }
    if (!A_F32 || !B_F32) wait_async();
    __syncthreads();

#pragma unroll
    for (int ks = 0; ks < 64; ks += 32) {
      v16bf af0 = frag_a_lds(&As[(wrow * 32 + 0 + lrow) * 64], ks, khalf);
      v16bf af1 = frag_a_lds(&As[(wrow * 32 + 16 + lrow) * 64], ks, khalf);
      v16bf bf0 = frag_b_lds(&Bs[(wcol * 32 + 0 + lrow) * 64], ks, khalf);
      v16bf bf1 = frag_b_lds(&Bs[(wcol * 32 + 16 + lrow) * 64], ks, khalf);
      acc[0][0] = wmma_bf16(af0, bf0, acc[0][0]);
      acc[0][1] = wmma_bf16(af0, bf1, acc[0][1]);
      acc[1][0] = wmma_bf16(af1, bf0, acc[1][0]);
      acc[1][1] = wmma_bf16(af1, bf1, acc[1][1]);
    }
    __syncthreads();
  }

#pragma unroll
  for (int ti = 0; ti < 2; ++ti) {
#pragma unroll
    for (int tj = 0; tj < 2; ++tj) {
      long long gn = bn + wcol * 32 + tj * 16 + lrow;
#pragma unroll
      for (int r = 0; r < 8; ++r) {
        long long gm = bm + wrow * 32 + ti * 16 + r + 8 * khalf;
        float val = acc[ti][tj][r] * alpha;
        if (bias) val += bias[gn];
        if (frame) {  // fused attn_frame @ Wfc[:, K:K+3]^T
          const float* fr = frame + gm * 3;
          const float* tb = tailB + gn * (long long)ldb + K;
          val += fr[0] * tb[0] + fr[1] * tb[1] + fr[2] * tb[2];
        }
        if (OUT_BF16)
          ((unsigned short*)Cv)[z * sC + gm * (long long)ldc + gn] = f2bf(val);
        else
          ((float*)Cv)[z * sC + gm * (long long)ldc + gn] = val;
      }
    }
  }
}

// ---------------------------------------------------------------------------
// Per-(b,h) attention: 32 query rows per block; full 32x1024 score strip lives
// in LDS (CDNA5: 320KB/WGP). scores = scale*Q*K^T (WMMA), softmax in LDS,
// O = P*V (WMMA, P converted bf16 on the fly, V transposed in LDS).
// Q/K tiles staged with async global->LDS copies.
// ---------------------------------------------------------------------------
__global__ __launch_bounds__(256) void attn_kernel(
    const unsigned short* __restrict__ Q, const unsigned short* __restrict__ Kb,
    const unsigned short* __restrict__ V, unsigned short* __restrict__ O, float scale) {
  extern __shared__ char smem[];
  float* S           = (float*)smem;                       // 32*1024 f32 = 128KB
  unsigned short* Qs = (unsigned short*)(S + 32 * 1024);   // 32*64 bf16
  unsigned short* KV = Qs + 32 * 64;                       // 64*64 bf16 (K tile, then V^T tile)
  float* red         = (float*)(KV + 64 * 64);             // 32*8
  float* red2        = red + 256;                          // 32*8

  const int t = threadIdx.x, lane = t & 31, w = t >> 5;
  const int lrow = lane & 15, khalf = lane >> 4;
  const int it = blockIdx.x, h = blockIdx.y, b = blockIdx.z;
  const int i0 = it * 32;
  const long long rowBase = (long long)b * SEQ;
  const unsigned short* Qp = Q + (rowBase + i0) * DMODEL + h * HDIM;
  const unsigned short* Kp = Kb + rowBase * DMODEL + h * HDIM;
  const unsigned short* Vp = V + rowBase * DMODEL + h * HDIM;

  // stage Q block 32x64 (async; waited together with first K tile)
  for (int c = t; c < 32 * 16; c += 256) {
    int r = c >> 4, col = (c & 15) * 4;
    async_copy_b64(&Qs[r * 64 + col], &Qp[(long long)r * DMODEL + col]);
  }

  const int iw = (w & 1) * 16;        // i sub-tile
  const int jw = (w >> 1) * 16;       // j (or d) sub-tile

  // ---- scores ----
  for (int jt = 0; jt < 16; ++jt) {
    for (int c = t; c < 64 * 16; c += 256) {  // stage K tile 64x64 (async)
      int r = c >> 4, col = (c & 15) * 4;
      async_copy_b64(&KV[r * 64 + col], &Kp[(long long)(jt * 64 + r) * DMODEL + col]);
    }
    wait_async();
    __syncthreads();
    v8f acc = {0.f, 0.f, 0.f, 0.f, 0.f, 0.f, 0.f, 0.f};
#pragma unroll
    for (int ks = 0; ks < 64; ks += 32) {
      v16bf a  = frag_a_lds(&Qs[(iw + lrow) * 64], ks, khalf);
      v16bf bb = frag_b_lds(&KV[(jw + lrow) * 64], ks, khalf);
      acc = wmma_bf16(a, bb, acc);
    }
#pragma unroll
    for (int r = 0; r < 8; ++r)
      S[(iw + r + 8 * khalf) * 1024 + jt * 64 + jw + lrow] = acc[r] * scale;
    __syncthreads();
  }

  // ---- softmax over S rows (8 threads per row) ----
  {
    const int row = t >> 3, sub = t & 7;
    float* Sr = S + row * 1024;
    float mx = -1e30f;
    for (int j = sub; j < 1024; j += 8) mx = fmaxf(mx, Sr[j]);
    red[row * 8 + sub] = mx;
    __syncthreads();
    float m2 = red[row * 8];
#pragma unroll
    for (int q = 1; q < 8; ++q) m2 = fmaxf(m2, red[row * 8 + q]);
    float sum = 0.f;
    for (int j = sub; j < 1024; j += 8) {
      float e = __expf(Sr[j] - m2);
      Sr[j] = e;
      sum += e;
    }
    red2[row * 8 + sub] = sum;
    __syncthreads();
    float s2 = 0.f;
#pragma unroll
    for (int q = 0; q < 8; ++q) s2 += red2[row * 8 + q];
    float inv = 1.0f / s2;
    for (int j = sub; j < 1024; j += 8) Sr[j] *= inv;
  }
  __syncthreads();

  // ---- O = P * V ----
  v8f oacc = {0.f, 0.f, 0.f, 0.f, 0.f, 0.f, 0.f, 0.f};
  for (int jt = 0; jt < 16; ++jt) {
    for (int c = t; c < 4096; c += 256) {  // stage V^T: KV[d][j] (transpose scatter)
      int d = c & 63, j = c >> 6;
      KV[d * 64 + j] = Vp[(long long)(jt * 64 + j) * DMODEL + d];
    }
    __syncthreads();
#pragma unroll
    for (int ks = 0; ks < 64; ks += 32) {
      FragU u;
      const float* Sr = S + (iw + lrow) * 1024 + jt * 64 + ks;
#pragma unroll
      for (int e = 0; e < 16; ++e) {
        int kk = (e & 7) + 16 * (e >> 3) + 8 * khalf;  // A-layout K index
        u.s[e] = f2bf(Sr[kk]);
      }
      v16bf bb = frag_b_lds(&KV[(jw + lrow) * 64], ks, khalf);
      oacc = wmma_bf16(u.v, bb, oacc);
    }
    __syncthreads();
  }
#pragma unroll
  for (int r = 0; r < 8; ++r) {
    long long gm = rowBase + i0 + iw + r + 8 * khalf;
    O[gm * DMODEL + h * HDIM + jw + lrow] = f2bf(oacc[r]);
  }
}

// ---------------------------------------------------------------------------
// v_3d = x @ W3^T + b3   (tiny: 6144 dot products)
// ---------------------------------------------------------------------------
__global__ void v3d_kernel(const float* __restrict__ x, const float* __restrict__ W3,
                           const float* __restrict__ b3, float* __restrict__ v3d) {
  int idx = blockIdx.x * blockDim.x + threadIdx.x;
  if (idx >= MTOT * 3) return;
  int row = idx / 3, n = idx % 3;
  const float* xr = x + (long long)row * DMODEL;
  const float* wr = W3 + (long long)n * DMODEL;
  float s = 0.f;
  for (int k = 0; k < DMODEL; ++k) s = fmaf(xr[k], wr[k], s);
  v3d[idx] = s + b3[n];
}

// ---------------------------------------------------------------------------
// Geometric frame: fused merged-softmax + column-sum-of-orientation trick +
// cross product + weighted mean over j. One block per (b,i); streams the
// 75.5MB orientation tensor exactly once (the memory-bound leg).
// ---------------------------------------------------------------------------
__global__ __launch_bounds__(256) void frame_kernel(
    const float* __restrict__ ssum, const float* __restrict__ ad,
    const float* __restrict__ orient, const float* __restrict__ v3d,
    float* __restrict__ frame) {
  __shared__ float red[256];
  const int t = threadIdx.x;
  const int bi = blockIdx.x;  // b*1024 + i
  const int b = bi >> 10;
  const float* Sr = ssum + (long long)bi * 1024;

  float mx = -1e30f;
  for (int j = t; j < 1024; j += 256) mx = fmaxf(mx, Sr[j]);
  red[t] = mx;
  __syncthreads();
  for (int o = 128; o > 0; o >>= 1) {
    if (t < o) red[t] = fmaxf(red[t], red[t + o]);
    __syncthreads();
  }
  mx = red[0];
  __syncthreads();
  float sum = 0.f;
  for (int j = t; j < 1024; j += 256) sum += __expf(Sr[j] - mx);
  red[t] = sum;
  __syncthreads();
  for (int o = 128; o > 0; o >>= 1) {
    if (t < o) red[t] += red[t + o];
    __syncthreads();
  }
  const float inv = 1.0f / red[0];
  __syncthreads();

  float a0 = 0.f, a1 = 0.f, a2 = 0.f;
  const long long base = (long long)bi * 1024;
  for (int j = t; j < 1024; j += 256) {
    if (j + 256 < 1024) __builtin_prefetch(orient + (base + j + 256) * 9, 0, 1);
    float wgt = __expf(Sr[j] - mx) * inv;
    const float* Or = orient + (base + j) * 9;
    const float* vj = v3d + (long long)(b * SEQ + j) * 3;
    // v3d_map[n] = (sum_m O[m][n]) * v3d[n]
    float m0 = (Or[0] + Or[3] + Or[6]) * vj[0];
    float m1 = (Or[1] + Or[4] + Or[7]) * vj[1];
    float m2 = (Or[2] + Or[5] + Or[8]) * vj[2];
    const float* A = ad + (base + j) * 3;
    float c0 = A[1] * m2 - A[2] * m1;
    float c1 = A[2] * m0 - A[0] * m2;
    float c2 = A[0] * m1 - A[1] * m0;
    a0 += wgt * c0; a1 += wgt * c1; a2 += wgt * c2;
  }
  red[t] = a0; __syncthreads();
  for (int o = 128; o > 0; o >>= 1) { if (t < o) red[t] += red[t + o]; __syncthreads(); }
  if (t == 0) frame[(long long)bi * 3 + 0] = red[0] * (1.0f / 1024.0f);
  __syncthreads();
  red[t] = a1; __syncthreads();
  for (int o = 128; o > 0; o >>= 1) { if (t < o) red[t] += red[t + o]; __syncthreads(); }
  if (t == 0) frame[(long long)bi * 3 + 1] = red[0] * (1.0f / 1024.0f);
  __syncthreads();
  red[t] = a2; __syncthreads();
  for (int o = 128; o > 0; o >>= 1) { if (t < o) red[t] += red[t + o]; __syncthreads(); }
  if (t == 0) frame[(long long)bi * 3 + 2] = red[0] * (1.0f / 1024.0f);
}

// ---------------------------------------------------------------------------
extern "C" void kernel_launch(void* const* d_in, const int* in_sizes, int n_in,
                              void* d_out, int out_size, void* d_ws, size_t ws_size,
                              hipStream_t stream) {
  (void)in_sizes; (void)n_in; (void)out_size; (void)ws_size;
  const float* x   = (const float*)d_in[0];
  const float* ad  = (const float*)d_in[1];
  const float* ori = (const float*)d_in[2];
  const float* Wq  = (const float*)d_in[3];
  const float* bq  = (const float*)d_in[4];
  const float* Wk  = (const float*)d_in[5];
  const float* bk  = (const float*)d_in[6];
  const float* Wv  = (const float*)d_in[7];
  const float* bv  = (const float*)d_in[8];
  const float* W3  = (const float*)d_in[9];
  const float* b3  = (const float*)d_in[10];
  const float* Wfc = (const float*)d_in[11];
  const float* bfc = (const float*)d_in[12];
  float* out = (float*)d_out;

  // workspace carve (~21 MB total)
  char* ws = (char*)d_ws;
  size_t off = 0;
  auto alloc = [&](size_t bytes) -> void* {
    void* p = ws + off;
    off = (off + bytes + 255) & ~(size_t)255;
    return p;
  };
  unsigned short* Qbf = (unsigned short*)alloc((size_t)MTOT * DMODEL * 2);
  unsigned short* Kbf = (unsigned short*)alloc((size_t)MTOT * DMODEL * 2);
  unsigned short* Vbf = (unsigned short*)alloc((size_t)MTOT * DMODEL * 2);
  unsigned short* Obf = (unsigned short*)alloc((size_t)MTOT * DMODEL * 2);
  float* ssum  = (float*)alloc((size_t)BATCH * SEQ * SEQ * 4);
  float* v3d   = (float*)alloc((size_t)MTOT * 3 * 4);
  float* frame = (float*)alloc((size_t)MTOT * 3 * 4);

  const float scale = 0.125f;  // 1/sqrt(64)

  // 1) QKV projections: [2048,768] x [768,768]^T -> bf16
  dim3 gP(DMODEL / 64, MTOT / 128, 1);
  gemm_nt_kernel<1, 1, 1><<<gP, 256, 0, stream>>>(x, DMODEL, 0, Wq, DMODEL, 0, Qbf, DMODEL, 0,
                                                  MTOT, DMODEL, DMODEL, 1.0f, bq, nullptr, nullptr);
  gemm_nt_kernel<1, 1, 1><<<gP, 256, 0, stream>>>(x, DMODEL, 0, Wk, DMODEL, 0, Kbf, DMODEL, 0,
                                                  MTOT, DMODEL, DMODEL, 1.0f, bk, nullptr, nullptr);
  gemm_nt_kernel<1, 1, 1><<<gP, 256, 0, stream>>>(x, DMODEL, 0, Wv, DMODEL, 0, Vbf, DMODEL, 0,
                                                  MTOT, DMODEL, DMODEL, 1.0f, bv, nullptr, nullptr);

  // 2) v_3d
  v3d_kernel<<<(MTOT * 3 + 255) / 256, 256, 0, stream>>>(x, W3, b3, v3d);

  // 3) merged scores: sum_h q_h k_h^T == Q_full K_full^T (heads partition D)
  dim3 gS(SEQ / 64, SEQ / 128, BATCH);
  gemm_nt_kernel<0, 0, 0><<<gS, 256, 0, stream>>>(
      Qbf, DMODEL, (long long)SEQ * DMODEL, Kbf, DMODEL, (long long)SEQ * DMODEL,
      ssum, SEQ, (long long)SEQ * SEQ, SEQ, SEQ, DMODEL, scale, nullptr, nullptr, nullptr);

  // 4) per-head attention (WMMA scores + softmax + WMMA PV)
  dim3 gA(SEQ / 32, NHEADS, BATCH);
  size_t smem = (size_t)32 * 1024 * 4 + 32 * 64 * 2 + 64 * 64 * 2 + 2048;
  attn_kernel<<<gA, 256, smem, stream>>>(Qbf, Kbf, Vbf, Obf, scale);

  // 5) geometric frame (streams orientation once)
  frame_kernel<<<MTOT, 256, 0, stream>>>(ssum, ad, ori, v3d, frame);

  // 6) output projection with fused bias + rank-3 frame tail
  dim3 gF(DMODEL / 64, MTOT / 128, 1);
  gemm_nt_kernel<0, 1, 0><<<gF, 256, 0, stream>>>(
      Obf, DMODEL, 0, Wfc, DMODEL + 3, 0, out, DMODEL, 0,
      MTOT, DMODEL, DMODEL, 1.0f, bfc, frame, Wfc);
}
